// LongcatFlashMLA_43963285242579
// MI455X (gfx1250) — matrix operand order repaired and checked by reading, not compiled, not run on hardware
//
#include <hip/hip_runtime.h>

// ---------------------------------------------------------------------------
// Types for CDNA5 WMMA (wave32, 16x16x32 bf16 -> f32)
// ---------------------------------------------------------------------------
typedef __bf16 bf16;
typedef __attribute__((ext_vector_type(8)))  __bf16 bf16x8;
typedef __attribute__((ext_vector_type(16))) __bf16 bf16x16;
typedef __attribute__((ext_vector_type(8)))  float  f32x8;

#define B_SZ   2
#define L_SZ   2048
#define HID_SZ 2048
#define H_SZ   16
#define QL_SZ  1536
#define KVL_SZ 512
#define ROPE_SZ 64
#define NOPE_SZ 128
#define VD_SZ  128
#define QKD_SZ 192

#define EPS_F    1e-5f
#define FQ_F     (1.0f / 12.0f)      // SQ * SCALE = sqrt(4/3) / sqrt(192)
#define SKV_F    2.0f                // sqrt(2048/512)
#define LN_THETA 9.210340371976184f  // ln(10000)

static __device__ __forceinline__ f32x8 wmma_bf16(bf16x16 a, bf16x16 b, f32x8 c) {
#if defined(__HIP_DEVICE_COMPILE__)
  return __builtin_amdgcn_wmma_f32_16x16x32_bf16(false, a, false, b, (short)0, c,
                                                 false, false);
#else
  (void)a; (void)b;
  return c;
#endif
}

static __device__ __forceinline__ bf16x16 load_frag2(const bf16* p0, const bf16* p1) {
  bf16x8 lo = *(const bf16x8*)p0;
  bf16x8 hi = *(const bf16x8*)p1;
  return __builtin_shufflevector(lo, hi, 0, 1, 2, 3, 4, 5, 6, 7,
                                 8, 9, 10, 11, 12, 13, 14, 15);
}

// CDNA5 async global->LDS copy of 16 bytes per lane (ASYNCcnt-tracked, no VGPR
// data round-trip). LDS byte address = low 32 bits of the generic pointer.
static __device__ __forceinline__ void async_copy16(bf16* lds_dst, const bf16* gsrc) {
#if defined(__HIP_DEVICE_COMPILE__)
  unsigned l = (unsigned)(unsigned long long)lds_dst;
  asm volatile("global_load_async_to_lds_b128 %0, %1, off"
               :: "v"(l), "v"(gsrc) : "memory");
#else
  *(bf16x8*)lds_dst = *(const bf16x8*)gsrc;
#endif
}

static __device__ __forceinline__ void async_wait0() {
#if defined(__HIP_DEVICE_COMPILE__)
  asm volatile("s_wait_asynccnt 0x0" ::: "memory");
#endif
}

// ---------------------------------------------------------------------------
// Elementwise conversion / prep kernels
// ---------------------------------------------------------------------------
__global__ void cvt_f32_bf16(const float* __restrict__ in, bf16* __restrict__ out,
                             long long n) {
  long long idx = (long long)blockIdx.x * blockDim.x + threadIdx.x;
  long long stride = (long long)gridDim.x * blockDim.x;
  for (long long i = idx; i < n; i += stride) out[i] = (bf16)in[i];
}

// w_embed_q (H, KVL, NOPE) f32 -> (H, NOPE, KVL) bf16  (so GEMM sees N x K)
__global__ void trans_wemb(const float* __restrict__ in, bf16* __restrict__ out) {
  long long idx = (long long)blockIdx.x * blockDim.x + threadIdx.x;
  if (idx >= (long long)H_SZ * KVL_SZ * NOPE_SZ) return;
  int h = (int)(idx >> 16);           // / (512*128)
  int rem = (int)(idx & 65535);
  int k = rem >> 7;                   // KVL index
  int n = rem & 127;                  // NOPE index
  out[(long long)h * (NOPE_SZ * KVL_SZ) + (long long)n * KVL_SZ + k] = (bf16)in[idx];
}

// rmsnorm of qa rows (len QL) -> bf16
__global__ __launch_bounds__(256) void rmsnorm_qa(const float* __restrict__ qa,
                                                  const float* __restrict__ w,
                                                  bf16* __restrict__ out) {
  __shared__ float sb[256];
  int row = blockIdx.x;
  const float* rp = qa + (long long)row * QL_SZ;
  float s = 0.f;
  for (int j = threadIdx.x; j < QL_SZ; j += 256) { float v = rp[j]; s += v * v; }
  sb[threadIdx.x] = s;
  __syncthreads();
  for (int o = 128; o > 0; o >>= 1) {
    if (threadIdx.x < o) sb[threadIdx.x] += sb[threadIdx.x + o];
    __syncthreads();
  }
  float rn = rsqrtf(sb[0] / (float)QL_SZ + EPS_F);
  bf16* op = out + (long long)row * QL_SZ;
  for (int j = threadIdx.x; j < QL_SZ; j += 256) op[j] = (bf16)(rp[j] * rn * w[j]);
}

// ckv rows (len 576): rmsnorm first 512 -> kv_latent bf16 (scaled by SKV),
// rope last 64 -> kpe bf16
__global__ __launch_bounds__(256) void prep_kv(const float* __restrict__ ckv,
                                               const float* __restrict__ w,
                                               bf16* __restrict__ kvlat,
                                               bf16* __restrict__ kpe) {
  __shared__ float sb[256];
  int row = blockIdx.x;
  int l = row & (L_SZ - 1);
  const float* cp = ckv + (long long)row * (KVL_SZ + ROPE_SZ);
  float v0 = cp[threadIdx.x], v1 = cp[threadIdx.x + 256];
  sb[threadIdx.x] = v0 * v0 + v1 * v1;
  __syncthreads();
  for (int o = 128; o > 0; o >>= 1) {
    if (threadIdx.x < o) sb[threadIdx.x] += sb[threadIdx.x + o];
    __syncthreads();
  }
  float rn = rsqrtf(sb[0] / (float)KVL_SZ + EPS_F);
  bf16* kp = kvlat + (long long)row * KVL_SZ;
  kp[threadIdx.x] = (bf16)(v0 * rn * w[threadIdx.x] * SKV_F);
  kp[threadIdx.x + 256] = (bf16)(v1 * rn * w[threadIdx.x + 256] * SKV_F);
  if (threadIdx.x < 32) {
    int i = threadIdx.x;
    float fr = __expf(-(2.0f * i / (float)ROPE_SZ) * LN_THETA);
    float th = (float)l * fr, c = __cosf(th), sn = __sinf(th);
    float x1 = cp[KVL_SZ + 2 * i], x2 = cp[KVL_SZ + 2 * i + 1];
    kpe[(long long)row * ROPE_SZ + 2 * i] = (bf16)(x1 * c - x2 * sn);
    kpe[(long long)row * ROPE_SZ + 2 * i + 1] = (bf16)(x1 * sn + x2 * c);
  }
}

// q (B*L, H*192) bf16 -> q_cat (B,H,L,192) bf16 with SQ*SCALE folded + rope on pe
__global__ __launch_bounds__(256) void prep_q(const bf16* __restrict__ q,
                                              bf16* __restrict__ qcat) {
  int row = blockIdx.x;                 // b*L + l
  int l = row & (L_SZ - 1);
  int b = row >> 11;
  const bf16* qrow = q + (long long)row * (H_SZ * QKD_SZ);
  for (int h = 0; h < H_SZ; ++h) {
    bf16* dst = qcat + (((long long)(b * H_SZ + h) * L_SZ) + l) * QKD_SZ;
    if (threadIdx.x < NOPE_SZ) {
      int d = threadIdx.x;
      dst[d] = (bf16)((float)qrow[h * QKD_SZ + d] * FQ_F);
    } else if (threadIdx.x < NOPE_SZ + 32) {
      int i = threadIdx.x - NOPE_SZ;
      float fr = __expf(-(2.0f * i / (float)ROPE_SZ) * LN_THETA);
      float th = (float)l * fr, c = __cosf(th), sn = __sinf(th);
      float x1 = (float)qrow[h * QKD_SZ + NOPE_SZ + 2 * i];
      float x2 = (float)qrow[h * QKD_SZ + NOPE_SZ + 2 * i + 1];
      dst[NOPE_SZ + 2 * i] = (bf16)((x1 * c - x2 * sn) * FQ_F);
      dst[NOPE_SZ + 2 * i + 1] = (bf16)((x1 * sn + x2 * c) * FQ_F);
    }
  }
}

// broadcast roped k_pe into k_cat[..., 128:192] for every head
__global__ void copy_kpe(const bf16* __restrict__ kpe, bf16* __restrict__ kcat) {
  long long idx = (long long)blockIdx.x * blockDim.x + threadIdx.x;
  if (idx >= (long long)B_SZ * H_SZ * L_SZ * ROPE_SZ) return;
  int j = (int)(idx & 63);
  int l = (int)((idx >> 6) & (L_SZ - 1));
  int h = (int)((idx >> 17) & 15);
  int b = (int)(idx >> 21);
  kcat[(((long long)(b * H_SZ + h) * L_SZ) + l) * QKD_SZ + NOPE_SZ + j] =
      kpe[((long long)(b * L_SZ + l)) * ROPE_SZ + j];
}

// ---------------------------------------------------------------------------
// Tiled GEMM: C(MxN) = A(MxK) * W(NxK)^T, bf16 operands, f32 accumulate.
// 256 threads = 8 waves; block tile 128x128, BK=32; wave tile 64x32.
// Double-buffered LDS, tiles staged with CDNA5 async global->LDS copies
// (issued before the WMMA chain of the current tile, waited after it).
// OOB rows are address-clamped: garbage only feeds discarded outputs.
// Batched over gridDim.z:  Aoff = (z/divA)*sA, Woff = (modB ? z%modB : z)*sB,
// Coff = z*sC.
// ---------------------------------------------------------------------------
#define LDT 40  // LDS row stride in halves (32 + 8 pad) -> 80B, 16B aligned

template <typename TOut>
__global__ __launch_bounds__(256) void gemm_bt(
    const bf16* __restrict__ A, const bf16* __restrict__ W, TOut* __restrict__ C,
    int M, int N, int K, int lda, int ldb, int ldc,
    long long sA, int divA, long long sB, int modB, long long sC) {
  __shared__ __align__(16) bf16 As[2][128 * LDT];
  __shared__ __align__(16) bf16 Bs[2][128 * LDT];

  int z = blockIdx.z;
  A += (long long)(z / divA) * sA;
  W += (long long)(modB ? (z % modB) : z) * sB;
  C += (long long)z * sC;

  const int tid = threadIdx.x;
  const int lane = tid & 31;
  const int wid = tid >> 5;
  const int ln = lane & 15;
  const int half = lane >> 4;
  const int m0 = blockIdx.y * 128;
  const int n0 = blockIdx.x * 128;
  const int wm = (wid >> 2) * 64;   // 2 wave rows of 64
  const int wn = (wid & 3) * 32;    // 4 wave cols of 32

  auto stage = [&](int buf, int k0) {
#pragma unroll
    for (int i = 0; i < 2; ++i) {
      int e = tid * 8 + i * 2048;
      int r = e >> 5, c = e & 31;
      int gm = m0 + r; if (gm >= M) gm = M - 1;
      async_copy16(&As[buf][r * LDT + c], A + (long long)gm * lda + k0 + c);
      int gn = n0 + r; if (gn >= N) gn = N - 1;
      async_copy16(&Bs[buf][r * LDT + c], W + (long long)gn * ldb + k0 + c);
    }
  };

  f32x8 acc[4][2];
#pragma unroll
  for (int mi = 0; mi < 4; ++mi)
#pragma unroll
    for (int ni = 0; ni < 2; ++ni) acc[mi][ni] = (f32x8)(0.0f);

  const int nk = K >> 5;
  stage(0, 0);
  async_wait0();
  __syncthreads();

  for (int kk = 0; kk < nk; ++kk) {
    const int cur = kk & 1;
    if (kk + 1 < nk) stage(1 - cur, (kk + 1) * 32);  // overlap with compute

    bf16x16 af[4];
#pragma unroll
    for (int mi = 0; mi < 4; ++mi) {
      const bf16* p = &As[cur][(wm + 16 * mi + ln) * LDT + 8 * half];
      af[mi] = load_frag2(p, p + 16);
    }
    bf16x16 bfr[2];
#pragma unroll
    for (int ni = 0; ni < 2; ++ni) {
      const bf16* p = &Bs[cur][(wn + 16 * ni + ln) * LDT + 8 * half];
      bfr[ni] = load_frag2(p, p + 16);
    }
#pragma unroll
    for (int mi = 0; mi < 4; ++mi)
#pragma unroll
      for (int ni = 0; ni < 2; ++ni)
        acc[mi][ni] = wmma_bf16(af[mi], bfr[ni], acc[mi][ni]);

    async_wait0();
    __syncthreads();
  }

#pragma unroll
  for (int mi = 0; mi < 4; ++mi)
#pragma unroll
    for (int ni = 0; ni < 2; ++ni) {
      int gn = n0 + wn + 16 * ni + ln;
      if (gn >= N) continue;
#pragma unroll
      for (int r = 0; r < 8; ++r) {
        int gm = m0 + wm + 16 * mi + r + 8 * half;
        if (gm < M) C[(long long)gm * ldc + gn] = (TOut)acc[mi][ni][r];
      }
    }
}

// ---------------------------------------------------------------------------
// Flash attention: one block per (128 queries, b, h). 8 waves, each owns 16
// queries; keys processed in 64-wide chunks (causal). K tile staged via async
// global->LDS copies; V staged transposed (manual); P round-trips through
// per-wave LDS for the C-layout -> A-layout change.
// ---------------------------------------------------------------------------
#define KLD 200  // K tile LDS row stride in halves (192 + 8)
#define VLD 72   // Vt / P LDS row stride in halves (64 + 8)

__global__ __launch_bounds__(256) void attn_mla(
    const bf16* __restrict__ qcat, const bf16* __restrict__ kcat,
    const bf16* __restrict__ vmat, bf16* __restrict__ attno) {
  __shared__ __align__(16) bf16 Ks[64 * KLD];
  __shared__ __align__(16) bf16 Vt[128 * VLD];
  __shared__ __align__(16) bf16 Ps[8 * 16 * VLD];

  const int tid = threadIdx.x;
  const int lane = tid & 31;
  const int wave = tid >> 5;
  const int ln = lane & 15;
  const int half = lane >> 4;
  const int qb = blockIdx.x * 128;
  const int bh = blockIdx.y;
  const int b = bh >> 4;
  const int h = bh & 15;

  const bf16* qp = qcat + ((long long)bh * L_SZ + qb) * QKD_SZ;
  const bf16* kp = kcat + (long long)bh * L_SZ * QKD_SZ;
  const bf16* vp = vmat + (long long)bh * L_SZ * VD_SZ;

  // Q fragments for this wave's 16 query rows: 6 k-steps of 32 over 192 dims.
  bf16x16 qf[6];
  {
    const bf16* qr = qp + (long long)(wave * 16 + ln) * QKD_SZ;
#pragma unroll
    for (int ks = 0; ks < 6; ++ks) {
      const bf16* p = qr + ks * 32 + 8 * half;
      qf[ks] = load_frag2(p, p + 16);
    }
  }

  float m_i[8], l_i[8];
  f32x8 Oa[8];
#pragma unroll
  for (int r = 0; r < 8; ++r) { m_i[r] = -1e30f; l_i[r] = 0.f; }
#pragma unroll
  for (int nt = 0; nt < 8; ++nt) Oa[nt] = (f32x8)(0.0f);

  const int nch = (qb >> 6) + 2;  // causal: keys up to qb+127
  for (int ch = 0; ch < nch; ++ch) {
    const int kv0 = ch * 64;
    __syncthreads();
    // stage K tile (64 x 192) row-major via async copies
#pragma unroll
    for (int i = 0; i < 6; ++i) {
      int e = i * 2048 + tid * 8;
      int r = e / QKD_SZ, c = e - r * QKD_SZ;
      async_copy16(&Ks[r * KLD + c], kp + (long long)(kv0 + r) * QKD_SZ + c);
    }
    // stage V tile transposed: Vt[vd][key]
#pragma unroll
    for (int i = 0; i < 4; ++i) {
      int e = i * 2048 + tid * 8;
      int key = e >> 7, vd = e & 127;
      bf16x8 v = *(const bf16x8*)(vp + (long long)(kv0 + key) * VD_SZ + vd);
#pragma unroll
      for (int j = 0; j < 8; ++j) Vt[(vd + j) * VLD + key] = v[j];
    }
    async_wait0();
    __syncthreads();

    // scores S (16 queries x 64 keys) as 4 WMMA C-tiles
    f32x8 S[4];
#pragma unroll
    for (int nt = 0; nt < 4; ++nt) {
      f32x8 a = (f32x8)(0.0f);
      const bf16* kr = Ks + (nt * 16 + ln) * KLD;
#pragma unroll
      for (int ks = 0; ks < 6; ++ks) {
        const bf16* p = kr + ks * 32 + 8 * half;
        a = wmma_bf16(qf[ks], load_frag2(p, p + 16), a);
      }
      int kpos = kv0 + nt * 16 + ln;
#pragma unroll
      for (int r = 0; r < 8; ++r) {
        int qpos = qb + wave * 16 + r + 8 * half;
        if (kpos > qpos) a[r] = -1e30f;
      }
      S[nt] = a;
    }

    // online softmax: row stats via 16-lane butterflies
    float mx[8];
#pragma unroll
    for (int r = 0; r < 8; ++r) {
      float m = fmaxf(fmaxf(S[0][r], S[1][r]), fmaxf(S[2][r], S[3][r]));
      m = fmaxf(m, __shfl_xor(m, 1, 32));
      m = fmaxf(m, __shfl_xor(m, 2, 32));
      m = fmaxf(m, __shfl_xor(m, 4, 32));
      m = fmaxf(m, __shfl_xor(m, 8, 32));
      mx[r] = m;
    }
    float alpha[8];
#pragma unroll
    for (int r = 0; r < 8; ++r) {
      float mn = fmaxf(m_i[r], mx[r]);
      alpha[r] = __expf(m_i[r] - mn);
      m_i[r] = mn;
    }
#pragma unroll
    for (int nt = 0; nt < 8; ++nt)
#pragma unroll
      for (int r = 0; r < 8; ++r) Oa[nt][r] *= alpha[r];

    float rs[8] = {0.f, 0.f, 0.f, 0.f, 0.f, 0.f, 0.f, 0.f};
#pragma unroll
    for (int nt = 0; nt < 4; ++nt)
#pragma unroll
      for (int r = 0; r < 8; ++r) {
        float p = __expf(S[nt][r] - m_i[r]);
        S[nt][r] = p;
        rs[r] += p;
        Ps[wave * 16 * VLD + (r + 8 * half) * VLD + nt * 16 + ln] = (bf16)p;
      }
#pragma unroll
    for (int r = 0; r < 8; ++r) {
      float s = rs[r];
      s += __shfl_xor(s, 1, 32);
      s += __shfl_xor(s, 2, 32);
      s += __shfl_xor(s, 4, 32);
      s += __shfl_xor(s, 8, 32);
      l_i[r] = l_i[r] * alpha[r] + s;
    }

    // P @ V : 2 k-steps of 32 keys, 8 n-tiles of 16 vd
    const bf16* pw = Ps + wave * 16 * VLD;
#pragma unroll
    for (int kst = 0; kst < 2; ++kst) {
      const bf16* pa = pw + ln * VLD + kst * 32 + 8 * half;
      bf16x16 ap = load_frag2(pa, pa + 16);
#pragma unroll
      for (int nt = 0; nt < 8; ++nt) {
        const bf16* vr = Vt + (nt * 16 + ln) * VLD + kst * 32 + 8 * half;
        Oa[nt] = wmma_bf16(ap, load_frag2(vr, vr + 16), Oa[nt]);
      }
    }
  }

  // epilogue: normalize and write attn_out[b, l, h*128 + vd] as bf16
  float inv[8];
#pragma unroll
  for (int r = 0; r < 8; ++r) inv[r] = 1.0f / l_i[r];
#pragma unroll
  for (int nt = 0; nt < 8; ++nt)
#pragma unroll
    for (int r = 0; r < 8; ++r) {
      int lrow = qb + wave * 16 + r + 8 * half;
      attno[((long long)b * L_SZ + lrow) * (H_SZ * VD_SZ) + h * VD_SZ + nt * 16 + ln] =
          (bf16)(Oa[nt][r] * inv[r]);
    }
}

// ---------------------------------------------------------------------------
// Host-side orchestration
// ---------------------------------------------------------------------------
extern "C" void kernel_launch(void* const* d_in, const int* in_sizes, int n_in,
                              void* d_out, int out_size, void* d_ws, size_t ws_size,
                              hipStream_t stream) {
  (void)in_sizes; (void)n_in; (void)out_size; (void)ws_size;
  const float* x        = (const float*)d_in[0];
  /* d_in[1] = mask (bool) — causal, handled analytically */
  const float* wq_a     = (const float*)d_in[2];
  const float* q_a_ln_w = (const float*)d_in[3];
  const float* wq_b     = (const float*)d_in[4];
  const float* wkv_a    = (const float*)d_in[5];
  const float* kv_a_ln_w= (const float*)d_in[6];
  const float* w_embed_q= (const float*)d_in[7];
  const float* w_unembed= (const float*)d_in[8];
  const float* wo       = (const float*)d_in[9];
  float* out = (float*)d_out;

  char* ws = (char*)d_ws;
  size_t off = 0;
  auto alloc = [&](size_t bytes) -> char* {
    off = (off + 255) & ~(size_t)255;
    char* p = ws + off;
    off += bytes;
    return p;
  };

  const long long NT  = (long long)B_SZ * L_SZ;            // 4096 token rows
  bf16* xb    = (bf16*)alloc(NT * HID_SZ * 2);
  bf16* wqab  = (bf16*)alloc((long long)QL_SZ * HID_SZ * 2);
  bf16* wqbb  = (bf16*)alloc((long long)H_SZ * QKD_SZ * QL_SZ * 2);
  bf16* wkvab = (bf16*)alloc((long long)(KVL_SZ + ROPE_SZ) * HID_SZ * 2);
  bf16* wembT = (bf16*)alloc((long long)H_SZ * NOPE_SZ * KVL_SZ * 2);
  bf16* wuneb = (bf16*)alloc((long long)H_SZ * VD_SZ * KVL_SZ * 2);
  bf16* wob   = (bf16*)alloc((long long)HID_SZ * H_SZ * VD_SZ * 2);
  float* qa   = (float*)alloc(NT * QL_SZ * 4);
  bf16* qan   = (bf16*)alloc(NT * QL_SZ * 2);
  bf16* qbm   = (bf16*)alloc(NT * H_SZ * QKD_SZ * 2);
  float* ckv  = (float*)alloc(NT * (KVL_SZ + ROPE_SZ) * 4);
  bf16* kvl   = (bf16*)alloc(NT * KVL_SZ * 2);
  bf16* kpe   = (bf16*)alloc(NT * ROPE_SZ * 2);
  bf16* qcat  = (bf16*)alloc((long long)B_SZ * H_SZ * L_SZ * QKD_SZ * 2);
  bf16* kcat  = (bf16*)alloc((long long)B_SZ * H_SZ * L_SZ * QKD_SZ * 2);
  bf16* vmat  = (bf16*)alloc((long long)B_SZ * H_SZ * L_SZ * VD_SZ * 2);
  bf16* attno = (bf16*)alloc(NT * H_SZ * VD_SZ * 2);

  dim3 blk(256);
  auto cvt = [&](const float* src, bf16* dst, long long n) {
    cvt_f32_bf16<<<4096, blk, 0, stream>>>(src, dst, n);
  };
  cvt(x, xb, NT * HID_SZ);
  cvt(wq_a, wqab, (long long)QL_SZ * HID_SZ);
  cvt(wq_b, wqbb, (long long)H_SZ * QKD_SZ * QL_SZ);
  cvt(wkv_a, wkvab, (long long)(KVL_SZ + ROPE_SZ) * HID_SZ);
  cvt(w_unembed, wuneb, (long long)H_SZ * VD_SZ * KVL_SZ);
  cvt(wo, wob, (long long)HID_SZ * H_SZ * VD_SZ);
  trans_wemb<<<(H_SZ * KVL_SZ * NOPE_SZ) / 256, blk, 0, stream>>>(w_embed_q, wembT);

  // qa = x @ wq_a^T   (4096 x 1536 x 2048) -> f32
  gemm_bt<float><<<dim3(QL_SZ / 128, NT / 128, 1), blk, 0, stream>>>(
      xb, wqab, qa, (int)NT, QL_SZ, HID_SZ, HID_SZ, HID_SZ, QL_SZ, 0, 1, 0, 0, 0);
  rmsnorm_qa<<<(int)NT, blk, 0, stream>>>(qa, q_a_ln_w, qan);

  // q = qan @ wq_b^T  (4096 x 3072 x 1536) -> bf16
  gemm_bt<bf16><<<dim3((H_SZ * QKD_SZ) / 128, NT / 128, 1), blk, 0, stream>>>(
      qan, wqbb, qbm, (int)NT, H_SZ * QKD_SZ, QL_SZ, QL_SZ, QL_SZ, H_SZ * QKD_SZ,
      0, 1, 0, 0, 0);

  // ckv = x @ wkv_a^T (4096 x 576 x 2048) -> f32
  gemm_bt<float><<<dim3((KVL_SZ + ROPE_SZ + 127) / 128, NT / 128, 1), blk, 0, stream>>>(
      xb, wkvab, ckv, (int)NT, KVL_SZ + ROPE_SZ, HID_SZ, HID_SZ, HID_SZ,
      KVL_SZ + ROPE_SZ, 0, 1, 0, 0, 0);

  prep_kv<<<(int)NT, blk, 0, stream>>>(ckv, kv_a_ln_w, kvl, kpe);
  prep_q<<<(int)NT, blk, 0, stream>>>(qbm, qcat);

  // per-(b,h): K = kv_latent @ w_embed_q  -> kcat[..., 0:128] (ldc = 192)
  gemm_bt<bf16><<<dim3(1, L_SZ / 128, B_SZ * H_SZ), blk, 0, stream>>>(
      kvl, wembT, kcat, L_SZ, NOPE_SZ, KVL_SZ, KVL_SZ, KVL_SZ, QKD_SZ,
      (long long)L_SZ * KVL_SZ, H_SZ, (long long)NOPE_SZ * KVL_SZ, H_SZ,
      (long long)L_SZ * QKD_SZ);

  // per-(b,h): V = kv_latent @ w_unembed^T -> vmat
  gemm_bt<bf16><<<dim3(1, L_SZ / 128, B_SZ * H_SZ), blk, 0, stream>>>(
      kvl, wuneb, vmat, L_SZ, VD_SZ, KVL_SZ, KVL_SZ, KVL_SZ, VD_SZ,
      (long long)L_SZ * KVL_SZ, H_SZ, (long long)VD_SZ * KVL_SZ, H_SZ,
      (long long)L_SZ * VD_SZ);

  copy_kpe<<<(B_SZ * H_SZ * L_SZ * ROPE_SZ) / 256, blk, 0, stream>>>(kpe, kcat);

  attn_mla<<<dim3(L_SZ / 128, B_SZ * H_SZ), blk, 0, stream>>>(qcat, kcat, vmat, attno);

  // out = attn_out @ wo^T (4096 x 2048 x 2048) -> f32 d_out
  gemm_bt<float><<<dim3(HID_SZ / 128, NT / 128, 1), blk, 0, stream>>>(
      attno, wob, out, (int)NT, HID_SZ, H_SZ * VD_SZ, H_SZ * VD_SZ, H_SZ * VD_SZ,
      HID_SZ, 0, 1, 0, 0, 0);
}
